// FNOBlock_37864431682095
// MI455X (gfx1250) — matrix operand order repaired and checked by reading, not compiled, not run on hardware
//
#include <hip/hip_runtime.h>
#include <math.h>

#define B_    16
#define CIN   128
#define COUT  128
#define NPTS  8192
#define MODES 64
#define KSPLIT 8
#define SA 40   // padded LDS row stride (elements); 80B = multiple of 16B

typedef __attribute__((ext_vector_type(16))) __bf16         bf16x16;
typedef __attribute__((ext_vector_type(8)))  float          f32x8;
typedef __attribute__((ext_vector_type(4)))  float          f32x4;
typedef __attribute__((ext_vector_type(4)))  unsigned int   u32x4;
typedef __attribute__((ext_vector_type(4)))  unsigned short u16x4;

struct U32x4Pair { u32x4 lo, hi; };

__device__ __forceinline__ unsigned short f2bf(float f) {
  unsigned int u = __builtin_bit_cast(unsigned int, f);
  u += 0x7FFFu + ((u >> 16) & 1u);
  return (unsigned short)(u >> 16);
}

// A fragment (16x32 bf16): lane L holds row M=L%16; lanes<16: K in {0..7,16..23},
// lanes>=16: K in {8..15,24..31}  (two 16B contiguous runs from LDS).
__device__ __forceinline__ bf16x16 frag_a(const unsigned short* s, int stride,
                                          int rowBase, int lane) {
  int r  = rowBase + (lane & 15);
  int kb = (lane >> 4) * 8;
  U32x4Pair p;
  p.lo = *reinterpret_cast<const u32x4*>(s + r * stride + kb);
  p.hi = *reinterpret_cast<const u32x4*>(s + r * stride + kb + 16);
  return __builtin_bit_cast(bf16x16, p);
}

// B fragment (32x16 bf16), LDS stored transposed [N][K]: lane L holds col N=L%16;
// lanes<16: K in {0..7,16..23}, lanes>=16: K in {8..15,24..31} (two 16B runs).
__device__ __forceinline__ bf16x16 frag_b(const unsigned short* s, int stride,
                                          int colBase, int lane) {
  int c  = colBase + (lane & 15);
  int kb = (lane >> 4) * 8;
  U32x4Pair p;
  p.lo = *reinterpret_cast<const u32x4*>(s + c * stride + kb);
  p.hi = *reinterpret_cast<const u32x4*>(s + c * stride + kb + 16);
  return __builtin_bit_cast(bf16x16, p);
}

__device__ __forceinline__ f32x8 wmma_bf16(bf16x16 a, bf16x16 b, f32x8 c) {
  return __builtin_amdgcn_wmma_f32_16x16x32_bf16(false, a, false, b,
                                                 (short)0, c, false, false);
}

// ---------------------------------------------------------------- basis init
// basisF: [c][n] (c=2m -> cos, c=2m+1 -> -sin), basisI: [n][c] with 1/N, 2/N scaling.
__global__ __launch_bounds__(256) void k_init_basis(unsigned short* __restrict__ basisF,
                                                    unsigned short* __restrict__ basisI) {
  const int idx = blockIdx.x * 256 + threadIdx.x;   // 0 .. 2^20-1
  {
    int c = idx >> 13;            // / 8192
    int n = idx & (NPTS - 1);
    int m = c >> 1;
    int ph = (m * n) & (NPTS - 1);
    float th = (float)ph * (6.28318530717958647692f / (float)NPTS);
    float sn, cs; __sincosf(th, &sn, &cs);
    basisF[idx] = f2bf((c & 1) ? -sn : cs);
  }
  {
    int n = idx >> 7;             // / 128
    int c = idx & 127;
    int m = c >> 1;
    int ph = (m * n) & (NPTS - 1);
    float th = (float)ph * (6.28318530717958647692f / (float)NPTS);
    float sn, cs; __sincosf(th, &sn, &cs);
    float alpha = ((m == 0) ? 1.0f : 2.0f) * (1.0f / (float)NPTS);
    basisI[idx] = f2bf((c & 1) ? (-alpha * sn) : (alpha * cs));
  }
}

// ------------------------------------------------------- GEMM1: truncated DFT
// C1p[ks][row=bi][c] (partial over K-split), rows=2048, cols=128, K=8192.
__global__ __launch_bounds__(256) void k_dft(const float* __restrict__ x,
                                             const unsigned short* __restrict__ basisF,
                                             float* __restrict__ C1p) {
  __shared__ __align__(16) unsigned short As[128 * SA];
  __shared__ __align__(16) unsigned short Bs[128 * SA];
  const int t = threadIdx.x, lane = t & 31, wave = t >> 5;
  const int m0 = blockIdx.x * 128;
  const int ks = blockIdx.y;
  const int kbase = ks * (NPTS / KSPLIT);
  const int kk0 = (t & 7) * 4;
  const int r0 = t >> 3;

  f32x8 acc[8] = {};

  for (int kt = 0; kt < (NPTS / KSPLIT); kt += 32) {
    const int k0 = kbase + kt;
#pragma unroll
    for (int q = 0; q < 4; ++q) {           // A tile: 128 rows x 32 K, f32 -> bf16
      int rr = r0 + 32 * q;
      f32x4 v = *reinterpret_cast<const f32x4*>(&x[(size_t)(m0 + rr) * NPTS + k0 + kk0]);
      u16x4 h; h.x = f2bf(v.x); h.y = f2bf(v.y); h.z = f2bf(v.z); h.w = f2bf(v.w);
      *reinterpret_cast<u16x4*>(&As[rr * SA + kk0]) = h;
    }
#pragma unroll
    for (int q = 0; q < 4; ++q) {           // B tile: 128 cols x 32 K, bf16 direct
      int cc = r0 + 32 * q;
      u16x4 h = *reinterpret_cast<const u16x4*>(&basisF[(size_t)cc * NPTS + k0 + kk0]);
      *reinterpret_cast<u16x4*>(&Bs[cc * SA + kk0]) = h;
    }
    __syncthreads();
    bf16x16 a = frag_a(As, SA, 16 * wave, lane);
#pragma unroll
    for (int j = 0; j < 8; ++j) {
      bf16x16 b = frag_b(Bs, SA, 16 * j, lane);
      acc[j] = wmma_bf16(a, b, acc[j]);
    }
    __syncthreads();
  }
  const int rl = (lane >> 4) * 8, cl = lane & 15;
#pragma unroll
  for (int j = 0; j < 8; ++j)
#pragma unroll
    for (int r = 0; r < 8; ++r) {
      int grow = m0 + 16 * wave + rl + r;
      int gcol = 16 * j + cl;
      C1p[((size_t)ks * 2048 + grow) * 128 + gcol] = acc[j][r];
    }
}

// ----------------------------------------------- deterministic K-split reduce
__global__ __launch_bounds__(256) void k_reduce(const float* __restrict__ C1p,
                                                float* __restrict__ C1) {
  const int idx = blockIdx.x * 256 + threadIdx.x;   // 0 .. 2048*128-1
  float s = 0.0f;
#pragma unroll
  for (int ks = 0; ks < KSPLIT; ++ks) s += C1p[(size_t)ks * (2048 * 128) + idx];
  C1[idx] = s;
}

// ------------------------------------------ GEMM2: per-mode complex mixing
// One block per mode m: A'[16 x 256] = [Xr | Xi], B'[256 x 256] packed complex,
// Y[b][c=2o+d] -> Ym[(b*128+o)*128 + 2m+d].
__global__ __launch_bounds__(128) void k_modes(const float* __restrict__ C1,
                                               const float* __restrict__ w_spec,
                                               float* __restrict__ Ym) {
  __shared__ __align__(16) unsigned short As[16 * SA];
  __shared__ __align__(16) unsigned short Bs[256 * SA];
  const int t = threadIdx.x, lane = t & 31, wave = t >> 5;   // 4 waves
  const int m = blockIdx.x;

  f32x8 acc[4] = {};

  for (int k0 = 0; k0 < 256; k0 += 32) {
    {
      int kk = t & 31;
      int kg = k0 + kk;
#pragma unroll
      for (int q = 0; q < 4; ++q) {
        int bb = (t >> 5) + 4 * q;
        float v = (kg < 128)
          ? C1[((size_t)bb * 128 + kg) * 128 + 2 * m]
          : C1[((size_t)bb * 128 + (kg - 128)) * 128 + 2 * m + 1];
        As[bb * SA + kk] = f2bf(v);
      }
    }
    for (int e = t; e < 256 * 32; e += 128) {
      int c = e >> 5, kk = e & 31;
      int kg = k0 + kk;
      int i = (kg < 128) ? kg : (kg - 128);
      int o = c >> 1, d = c & 1;
      const float* wp = &w_spec[(((size_t)i * 128 + o) * 64 + m) * 2];
      float v = (kg < 128) ? (d ? wp[1] : wp[0]) : (d ? wp[0] : -wp[1]);
      Bs[c * SA + kk] = f2bf(v);
    }
    __syncthreads();
    bf16x16 a = frag_a(As, SA, 0, lane);
#pragma unroll
    for (int j = 0; j < 4; ++j) {
      bf16x16 b = frag_b(Bs, SA, 64 * wave + 16 * j, lane);
      acc[j] = wmma_bf16(a, b, acc[j]);
    }
    __syncthreads();
  }
  const int rl = (lane >> 4) * 8, cl = lane & 15;
#pragma unroll
  for (int j = 0; j < 4; ++j)
#pragma unroll
    for (int r = 0; r < 8; ++r) {
      int bb = rl + r;                      // batch 0..15
      int c  = 64 * wave + 16 * j + cl;     // 0..255
      int o = c >> 1, d = c & 1;
      Ym[((size_t)bb * 128 + o) * 128 + 2 * m + d] = acc[j][r];
    }
}

// ------------------- GEMM3 fused: out = gelu(Wconv*X_b + Ym_b*Bi^T + bias)
__global__ __launch_bounds__(256) void k_fused_out(const float* __restrict__ x,
                                                   const float* __restrict__ w_conv,
                                                   const float* __restrict__ b_conv,
                                                   const float* __restrict__ Ym,
                                                   const unsigned short* __restrict__ basisI,
                                                   float* __restrict__ out) {
  __shared__ __align__(16) unsigned short As[128 * SA];
  __shared__ __align__(16) unsigned short Bs[128 * SA];
  const int t = threadIdx.x, lane = t & 31, wave = t >> 5;
  const int n0 = blockIdx.x * 128;
  const int b  = blockIdx.y;
  const int kk0 = (t & 7) * 4;
  const int r0 = t >> 3;

  f32x8 acc[8] = {};

  // Phase 1: pointwise conv, K = Cin
  for (int k0 = 0; k0 < 128; k0 += 32) {
#pragma unroll
    for (int q = 0; q < 4; ++q) {
      int oo = r0 + 32 * q;
      f32x4 v = *reinterpret_cast<const f32x4*>(&w_conv[(size_t)oo * 128 + k0 + kk0]);
      u16x4 h; h.x = f2bf(v.x); h.y = f2bf(v.y); h.z = f2bf(v.z); h.w = f2bf(v.w);
      *reinterpret_cast<u16x4*>(&As[oo * SA + kk0]) = h;
    }
    {
      int j0 = (t & 31) * 4;                 // transpose-stage x[b][i][n] -> Bs[n][i]
#pragma unroll
      for (int q = 0; q < 4; ++q) {
        int kk = (t >> 5) + 8 * q;
        f32x4 v = *reinterpret_cast<const f32x4*>(
            &x[((size_t)b * 128 + k0 + kk) * NPTS + n0 + j0]);
        Bs[(j0 + 0) * SA + kk] = f2bf(v.x);
        Bs[(j0 + 1) * SA + kk] = f2bf(v.y);
        Bs[(j0 + 2) * SA + kk] = f2bf(v.z);
        Bs[(j0 + 3) * SA + kk] = f2bf(v.w);
      }
    }
    __syncthreads();
    bf16x16 a = frag_a(As, SA, 16 * wave, lane);
#pragma unroll
    for (int j = 0; j < 8; ++j) {
      bf16x16 bb = frag_b(Bs, SA, 16 * j, lane);
      acc[j] = wmma_bf16(a, bb, acc[j]);
    }
    __syncthreads();
  }
  // Phase 2: inverse DFT, K = 128 spectral coeffs
  for (int k0 = 0; k0 < 128; k0 += 32) {
#pragma unroll
    for (int q = 0; q < 4; ++q) {
      int oo = r0 + 32 * q;
      f32x4 v = *reinterpret_cast<const f32x4*>(&Ym[((size_t)b * 128 + oo) * 128 + k0 + kk0]);
      u16x4 h; h.x = f2bf(v.x); h.y = f2bf(v.y); h.z = f2bf(v.z); h.w = f2bf(v.w);
      *reinterpret_cast<u16x4*>(&As[oo * SA + kk0]) = h;
    }
#pragma unroll
    for (int q = 0; q < 4; ++q) {
      int jj = r0 + 32 * q;
      u16x4 h = *reinterpret_cast<const u16x4*>(&basisI[(size_t)(n0 + jj) * 128 + k0 + kk0]);
      *reinterpret_cast<u16x4*>(&Bs[jj * SA + kk0]) = h;
    }
    __syncthreads();
    bf16x16 a = frag_a(As, SA, 16 * wave, lane);
#pragma unroll
    for (int j = 0; j < 8; ++j) {
      bf16x16 bb = frag_b(Bs, SA, 16 * j, lane);
      acc[j] = wmma_bf16(a, bb, acc[j]);
    }
    __syncthreads();
  }
  // Epilogue: bias + exact GELU
  const int rl = (lane >> 4) * 8, cl = lane & 15;
#pragma unroll
  for (int j = 0; j < 8; ++j)
#pragma unroll
    for (int r = 0; r < 8; ++r) {
      int o = 16 * wave + rl + r;
      int n = n0 + 16 * j + cl;
      float z = acc[j][r] + b_conv[o];
      float g = 0.5f * z * (1.0f + erff(z * 0.70710678118654752f));
      out[((size_t)b * 128 + o) * NPTS + n] = g;
    }
}

extern "C" void kernel_launch(void* const* d_in, const int* in_sizes, int n_in,
                              void* d_out, int out_size, void* d_ws, size_t ws_size,
                              hipStream_t stream) {
  const float* x      = (const float*)d_in[0];   // [16,128,8192]
  const float* w_spec = (const float*)d_in[1];   // [128,128,64,2]
  const float* w_conv = (const float*)d_in[2];   // [128,128]
  const float* b_conv = (const float*)d_in[3];   // [128]
  float* out = (float*)d_out;                    // [16,128,8192]

  char* ws = (char*)d_ws;
  unsigned short* basisF = (unsigned short*)(ws);                       // 2 MB [128][8192]
  unsigned short* basisI = (unsigned short*)(ws + ((size_t)2 << 20));   // 2 MB [8192][128]
  float* C1p = (float*)(ws + ((size_t)4 << 20));                        // 8 MB partials
  float* C1  = (float*)(ws + ((size_t)12 << 20));                       // 1 MB [2048][128]
  float* Ym  = (float*)(ws + ((size_t)13 << 20));                       // 1 MB [2048][128]

  k_init_basis<<<4096, 256, 0, stream>>>(basisF, basisI);
  k_dft<<<dim3(16, KSPLIT), 256, 0, stream>>>(x, basisF, C1p);
  k_reduce<<<1024, 256, 0, stream>>>(C1p, C1);
  k_modes<<<MODES, 128, 0, stream>>>(C1, w_spec, Ym);
  k_fused_out<<<dim3(NPTS / 128, B_), 256, 0, stream>>>(x, w_conv, b_conv, Ym, basisI, out);
}
